// LSTMStack_47828755808362
// MI455X (gfx1250) — matrix operand order repaired and checked by reading, not compiled, not run on hardware
//
#include <hip/hip_runtime.h>
#include <stdint.h>

// ---------------------------------------------------------------------------
// 2-layer LSTM (B=32, T=512, I=H=1024) for gfx1250 (MI455X, wave32, WMMA)
//   1) convert x / weights to bf16 (fp32 accumulation everywhere)
//   2) gx = X @ W_ih^T + (b_ih + b_hh)   -- big parallel bf16 WMMA GEMM
//   3) persistent recurrence kernel: W_hh resident in LDS (staged by TDM),
//      per-step 16x16x32 bf16 WMMA + fused LSTM cell, grid barrier per step,
//      next step's gx tile prefetched into LDS by TDM (double buffered,
//      hidden behind WMMA compute, tracked with TENSORcnt)
// ---------------------------------------------------------------------------

typedef __attribute__((ext_vector_type(16))) __bf16 v16bf;
typedef __attribute__((ext_vector_type(8)))  __bf16 v8bf;
typedef __attribute__((ext_vector_type(8)))  float  v8f;

#define B_   32
#define T_   512
#define H_   1024
#define G4_  4096            // 4*H
#define NWG_ 64              // recurrence workgroups (each owns 16 h-columns)

// fp32 -> bf16, round-to-nearest-even
__device__ __forceinline__ __bf16 f2bf(float f) {
  union { float f; uint32_t u; } v; v.f = f;
  uint32_t r = v.u + 0x7FFFu + ((v.u >> 16) & 1u);
  uint16_t h = (uint16_t)(r >> 16);
  __bf16 o; __builtin_memcpy(&o, &h, 2);
  return o;
}

__device__ __forceinline__ float sigmoidf_(float x) {
  return 1.0f / (1.0f + __expf(-x));
}

// ---------------------------------------------------------------------------
// Tensor Data Mover support (gfx1250). Descriptor bitfields per CDNA5 ISA §8.
// ---------------------------------------------------------------------------
#if __has_builtin(__builtin_amdgcn_tensor_load_to_lds)
#define HAVE_TDM 1
typedef unsigned int u32x4 __attribute__((ext_vector_type(4)));
typedef int          i32x4 __attribute__((ext_vector_type(4)));
typedef int          i32x8 __attribute__((ext_vector_type(8)));

__device__ __forceinline__ void tdm_load(u32x4 g0, i32x8 g1, i32x4 g2, i32x4 g3) {
#if __clang_major__ >= 23
  i32x8 g4 = {0, 0, 0, 0, 0, 0, 0, 0};
  __builtin_amdgcn_tensor_load_to_lds(g0, g1, g2, g3, g4, 0);
#else
  __builtin_amdgcn_tensor_load_to_lds(g0, g1, g2, g3, 0);
#endif
}

// byte offset of a generic pointer within this WG's LDS allocation
__device__ __forceinline__ unsigned lds_offset(const void* p) {
  return (unsigned)(unsigned long long)
      (__attribute__((address_space(3))) const char*)(const char*)p;
}

// D# group0: count=1 | lds_addr | global_addr[56:0] | type=2
__device__ __forceinline__ u32x4 make_g0(unsigned lds_addr, unsigned long long ga) {
  u32x4 g;
  g.x = 1u;
  g.y = lds_addr;
  g.z = (unsigned)(ga & 0xFFFFFFFFu);
  g.w = (unsigned)((ga >> 32) & 0x01FFFFFFu) | 0x80000000u;
  return g;
}

// W_hh slice: 2-D tile, line = 16 contiguous rows (16*1024 bf16), 4 lines
// (gates) with line stride 1024 rows. data_size=2B.
__device__ __forceinline__ void tdm_issue_whh(const __bf16* Whh, int hb,
                                              unsigned lds) {
  unsigned long long ga = (unsigned long long)(size_t)(Whh + (size_t)hb * H_);
  u32x4 g0 = make_g0(lds, ga);
  i32x8 g1 = {0x00010000,   // data_size=1 (2 bytes)
              0x00000000,   // tensor_dim0 lo16 (dim0 = 16M)
              0x00040100,   // tensor_dim0 hi16 = 0x0100 | tensor_dim1(4)<<16
              0x40000000,   // tile_dim0 = 16384 elements << 16
              4,            // tile_dim1 = 4
              0x00100000,   // tensor_dim0_stride = 1048576 elements (lo32)
              0, 0};
  i32x4 z = {0, 0, 0, 0};
  tdm_load(g0, g1, z, z);
}

// per-step gx tile: 3-D tile, x = 16 cols (f32), y = 32 rows stride 4096,
// z = 4 gates stride 1024. LDS layout -> [gate][b][16] floats.
__device__ __forceinline__ void tdm_issue_gx(const float* gx, int t, int hb,
                                             unsigned lds) {
  unsigned long long ga =
      (unsigned long long)(size_t)(gx + (size_t)t * B_ * G4_ + hb);
  u32x4 g0 = make_g0(lds, ga);
  i32x8 g1 = {0x00020000,   // data_size=2 (4 bytes)
              0x10000000,   // tensor_dim0 = 4096 (lo16 << 16)
              0x00200000,   // tensor_dim1 = 32 (<<16)
              0x00100000,   // tile_dim0 = 16 (<<16)
              0x00040020,   // tile_dim1 = 32 | tile_dim2 = 4 << 16
              4096,         // tensor_dim0_stride = 4096
              0x04000000,   // tensor_dim1_stride = 1024 (<<16)
              0};
  i32x4 g2 = {4, 0, 0, 0};  // tensor_dim2 = 4
  i32x4 g3 = {0, 0, 0, 0};
  tdm_load(g0, g1, g2, g3);
}
#endif

__device__ __forceinline__ void wait_tensorcnt0() {
#if __has_builtin(__builtin_amdgcn_s_wait_tensorcnt)
  __builtin_amdgcn_s_wait_tensorcnt(0);
#else
  asm volatile("s_wait_tensorcnt 0x0" ::: "memory");
#endif
}

// ---------------------------------------------------------------------------
// conversion kernels
// ---------------------------------------------------------------------------
// x[B][T][I] (fp32) -> xb[T][B][I] (bf16)
__global__ void k_convert_x(const float* __restrict__ x, __bf16* __restrict__ xb) {
  const size_t total = (size_t)B_ * T_ * H_;
  for (size_t i = (size_t)blockIdx.x * blockDim.x + threadIdx.x; i < total;
       i += (size_t)gridDim.x * blockDim.x) {
    size_t b = i / ((size_t)T_ * H_);
    size_t r = i % ((size_t)T_ * H_);
    size_t t = r / H_, d = r % H_;
    xb[(t * B_ + b) * H_ + d] = f2bf(x[i]);
  }
}

__global__ void k_convert_w(const float* __restrict__ w, __bf16* __restrict__ wb, int n) {
  for (int i = blockIdx.x * blockDim.x + threadIdx.x; i < n;
       i += gridDim.x * blockDim.x)
    wb[i] = f2bf(w[i]);
}

__global__ void k_bias_sum(const float* __restrict__ bi, const float* __restrict__ bh,
                           float* __restrict__ o) {
  int i = blockIdx.x * blockDim.x + threadIdx.x;
  if (i < G4_) o[i] = bi[i] + bh[i];
}

// ---------------------------------------------------------------------------
// gx = A @ W^T + bias      A:[T*B,H] bf16, W:[4H,H] bf16, out:[T*B,4H] f32
// grid: (4096/64, T), block 256 (8 waves); wave w -> (m-half = w&1, n-tile = w>>1)
// ---------------------------------------------------------------------------
__global__ void k_gates_gemm(const __bf16* __restrict__ A,
                             const __bf16* __restrict__ W,
                             const float*  __restrict__ bias,
                             float* __restrict__ out) {
  const int t     = blockIdx.y;
  const int nb    = blockIdx.x * 64;
  const int w     = threadIdx.x >> 5;
  const int lane  = threadIdx.x & 31;
  const int mh    = w & 1;
  const int n0    = nb + (w >> 1) * 16;
  const int col   = lane & 15;
  const int khalf = lane >> 4;

  const __bf16* aRow = A + ((size_t)t * B_ + mh * 16 + col) * H_;
  const __bf16* bRow = W + (size_t)(n0 + col) * H_;

  v8f acc;
  const float bv = bias[n0 + col];
#pragma unroll
  for (int r = 0; r < 8; ++r) acc[r] = bv;

#pragma unroll 4
  for (int kk = 0; kk < H_ / 32; ++kk) {
    const int kb = kk * 32;
    v8bf a0 = *(const v8bf*)(aRow + kb + 8 * khalf);
    v8bf a1 = *(const v8bf*)(aRow + kb + 8 * khalf + 16);
    v8bf b0 = *(const v8bf*)(bRow + kb + 8 * khalf);
    v8bf b1 = *(const v8bf*)(bRow + kb + 8 * khalf + 16);
    v16bf av = __builtin_shufflevector(a0, a1, 0,1,2,3,4,5,6,7,8,9,10,11,12,13,14,15);
    v16bf bvv = __builtin_shufflevector(b0, b1, 0,1,2,3,4,5,6,7,8,9,10,11,12,13,14,15);
    acc = __builtin_amdgcn_wmma_f32_16x16x32_bf16(false, av, false, bvv,
                                                  (short)0, acc, false, false);
  }

  float* oBase = out + (size_t)t * B_ * G4_;
#pragma unroll
  for (int r = 0; r < 8; ++r) {
    const int m = mh * 16 + r + 8 * khalf;
    oBase[(size_t)m * G4_ + n0 + col] = acc[r];
  }
}

// ---------------------------------------------------------------------------
// grid-wide barrier: monotonic arrival counter, no reset race
// ---------------------------------------------------------------------------
__device__ __forceinline__ void grid_sync(unsigned* cnt, unsigned target) {
  __syncthreads();
  if (threadIdx.x == 0) {
    __threadfence();
    __hip_atomic_fetch_add(cnt, 1u, __ATOMIC_RELEASE, __HIP_MEMORY_SCOPE_AGENT);
    while (__hip_atomic_load(cnt, __ATOMIC_ACQUIRE, __HIP_MEMORY_SCOPE_AGENT) < target)
      __builtin_amdgcn_s_sleep(2);
  }
  __syncthreads();
}

// ---------------------------------------------------------------------------
// persistent LSTM recurrence.
// block x = h-slice (16 columns); 8 waves: (m-half = w&1, gate = w>>1)
// LDS: W_hh slice 64x1024 bf16 (128KB, TDM-staged) + gate exchange 8KB +
//      c state 2KB + double-buffered gx tile 2x8KB (TDM-prefetched)
// ---------------------------------------------------------------------------
#define GATE_OFF (64 * H_ * 2)
#define C_OFF    (GATE_OFF + 8 * 256 * 4)
#define GXT_OFF  (C_OFF + B_ * 16 * 4)
#define SMEM_REC (GXT_OFF + 2 * 4 * B_ * 16 * 4)

__global__ void k_lstm_rec(const float*  __restrict__ gx,    // [T*B,4H]
                           const __bf16* __restrict__ Whh,   // [4H,H]
                           __bf16* __restrict__ hbuf,        // [2,B,H]
                           __bf16* __restrict__ y16,         // [T*B,H] or null
                           float*  __restrict__ yf32,        // [T*B,H] or null
                           float*  __restrict__ hT,          // [B,H]
                           float*  __restrict__ cT,          // [B,H]
                           unsigned* __restrict__ cnt) {
  extern __shared__ char smem[];
  __bf16* w_lds    = (__bf16*)smem;                  // 64 x 1024
  float*  gate_lds = (float*)(smem + GATE_OFF);      // 8 x 16 x 16
  float*  c_lds    = (float*)(smem + C_OFF);         // 32 x 16
  float*  gxt      = (float*)(smem + GXT_OFF);       // 2 x [4][32][16]

  const int hb    = blockIdx.x * 16;
  const int tid   = threadIdx.x;
  const int w     = tid >> 5;
  const int lane  = tid & 31;
  const int mh    = w & 1;
  const int g     = w >> 1;
  const int col   = lane & 15;
  const int khalf = lane >> 4;

#if HAVE_TDM
  // stage W_hh slice + first gx tile via Tensor Data Mover
  if (w == 0) {
    tdm_issue_whh(Whh, hb, lds_offset(w_lds));
    tdm_issue_gx(gx, 0, hb, lds_offset(gxt));
  }
#else
  // manual stage of W_hh slice (row gate*16+c <- Whh[gate*1024+hb+c])
  for (int idx = tid; idx < 64 * (H_ / 2); idx += blockDim.x) {
    const int row = idx >> 9, d = idx & 511;
    const int gate = row >> 4, c = row & 15;
    ((uint32_t*)w_lds)[idx] =
        ((const uint32_t*)(Whh + (size_t)(gate * H_ + hb + c) * H_))[d];
  }
#endif
  for (int e = tid; e < B_ * 16; e += blockDim.x) c_lds[e] = 0.0f;
  __syncthreads();

  const __bf16* wRow = w_lds + (size_t)(g * 16 + col) * H_;
  int p = 0;

  for (int t = 0; t < T_; ++t) {
#if HAVE_TDM
    if (w == 0) wait_tensorcnt0();     // gx tile t (and W_hh at t=0) landed
    __syncthreads();
    if (w == 0 && t + 1 < T_)          // prefetch next step behind the WMMAs
      tdm_issue_gx(gx, t + 1, hb, lds_offset(gxt) + (unsigned)((p ^ 1) * 4 * B_ * 16 * 4));
#endif

    const __bf16* hcur = hbuf + (size_t)(t & 1) * B_ * H_;
    const __bf16* aRow = hcur + (size_t)(mh * 16 + col) * H_;

    v8f acc;
#if HAVE_TDM
    {
      const float* gt = gxt + (size_t)p * 4 * B_ * 16 + (size_t)g * B_ * 16;
#pragma unroll
      for (int r = 0; r < 8; ++r)
        acc[r] = gt[(mh * 16 + r + 8 * khalf) * 16 + col];
    }
#else
    {
      const float* gBase = gx + (size_t)t * B_ * G4_;
      const int n = g * H_ + hb + col;
#pragma unroll
      for (int r = 0; r < 8; ++r)
        acc[r] = gBase[(size_t)(mh * 16 + r + 8 * khalf) * G4_ + n];
    }
#endif

#pragma unroll 4
    for (int kk = 0; kk < H_ / 32; ++kk) {
      const int kb = kk * 32;
      v8bf a0 = *(const v8bf*)(aRow + kb + 8 * khalf);
      v8bf a1 = *(const v8bf*)(aRow + kb + 8 * khalf + 16);
      v8bf b0 = *(const v8bf*)(wRow + kb + 8 * khalf);
      v8bf b1 = *(const v8bf*)(wRow + kb + 8 * khalf + 16);
      v16bf av = __builtin_shufflevector(a0, a1, 0,1,2,3,4,5,6,7,8,9,10,11,12,13,14,15);
      v16bf bv = __builtin_shufflevector(b0, b1, 0,1,2,3,4,5,6,7,8,9,10,11,12,13,14,15);
      acc = __builtin_amdgcn_wmma_f32_16x16x32_bf16(false, av, false, bv,
                                                    (short)0, acc, false, false);
    }

#pragma unroll
    for (int r = 0; r < 8; ++r)
      gate_lds[w * 256 + (r + 8 * khalf) * 16 + col] = acc[r];
    __syncthreads();

    // fused LSTM cell for this WG's 32 x 16 (b, h) slice
    __bf16* hnext = hbuf + (size_t)((t + 1) & 1) * B_ * H_;
    for (int e = tid; e < B_ * 16; e += blockDim.x) {
      const int b = e >> 4, hc = e & 15;
      const int mh2 = b >> 4, ml = b & 15;
      const float iv = gate_lds[(0 + mh2) * 256 + ml * 16 + hc];
      const float fv = gate_lds[(2 + mh2) * 256 + ml * 16 + hc];
      const float gv = gate_lds[(4 + mh2) * 256 + ml * 16 + hc];
      const float ov = gate_lds[(6 + mh2) * 256 + ml * 16 + hc];
      const float cn = sigmoidf_(fv) * c_lds[e] + sigmoidf_(iv) * tanhf(gv);
      const float hn = sigmoidf_(ov) * tanhf(cn);
      c_lds[e] = cn;
      const size_t hidx = (size_t)b * H_ + hb + hc;
      hnext[hidx] = f2bf(hn);
      if (y16)  y16 [(size_t)t * B_ * H_ + hidx] = f2bf(hn);
      if (yf32) yf32[(size_t)t * B_ * H_ + hidx] = hn;
      if (t == T_ - 1) { hT[hidx] = hn; cT[hidx] = cn; }
    }
    grid_sync(cnt, (unsigned)((t + 1) * NWG_));
    p ^= 1;
  }
}

// ---------------------------------------------------------------------------
// host launcher
// ---------------------------------------------------------------------------
extern "C" void kernel_launch(void* const* d_in, const int* in_sizes, int n_in,
                              void* d_out, int out_size, void* d_ws, size_t ws_size,
                              hipStream_t stream) {
  (void)in_sizes; (void)n_in; (void)out_size; (void)ws_size;

  const float* x    = (const float*)d_in[0];
  const float* Wih0 = (const float*)d_in[1];
  const float* Whh0 = (const float*)d_in[2];
  const float* bih0 = (const float*)d_in[3];
  const float* bhh0 = (const float*)d_in[4];
  const float* Wih1 = (const float*)d_in[5];
  const float* Whh1 = (const float*)d_in[6];
  const float* bih1 = (const float*)d_in[7];
  const float* bhh1 = (const float*)d_in[8];
  float* out = (float*)d_out;

  // workspace layout (bytes)
  constexpr size_t XB   = 0;                                   // [T,B,H] bf16   32MB
  constexpr size_t Y0   = XB + (size_t)T_ * B_ * H_ * 2;       // [T,B,H] bf16   32MB
  constexpr size_t WI0  = Y0 + (size_t)T_ * B_ * H_ * 2;       // bf16 weights    8MB each
  constexpr size_t WH0  = WI0 + (size_t)G4_ * H_ * 2;
  constexpr size_t WI1  = WH0 + (size_t)G4_ * H_ * 2;
  constexpr size_t WH1  = WI1 + (size_t)G4_ * H_ * 2;
  constexpr size_t BI0  = WH1 + (size_t)G4_ * H_ * 2;          // fused bias f32
  constexpr size_t BI1  = BI0 + (size_t)G4_ * 4;
  constexpr size_t HB   = BI1 + (size_t)G4_ * 4;               // h double buffer bf16
  constexpr size_t CNT  = HB + (size_t)2 * B_ * H_ * 2;        // barrier counter
  constexpr size_t GX   = (CNT + 256 + 255) & ~(size_t)255;    // [T,B,4H] f32  256MB

  char* ws = (char*)d_ws;
  __bf16* xb    = (__bf16*)(ws + XB);
  __bf16* y0b   = (__bf16*)(ws + Y0);
  __bf16* wi0b  = (__bf16*)(ws + WI0);
  __bf16* wh0b  = (__bf16*)(ws + WH0);
  __bf16* wi1b  = (__bf16*)(ws + WI1);
  __bf16* wh1b  = (__bf16*)(ws + WH1);
  float*  b0    = (float*)(ws + BI0);
  float*  b1    = (float*)(ws + BI1);
  __bf16* hbuf  = (__bf16*)(ws + HB);
  unsigned* cnt = (unsigned*)(ws + CNT);
  float*  gxws  = (float*)(ws + GX);

  // d_out layout: outputs [T,B,H] | h-stack [2,B,H] | c-stack [2,B,H]
  constexpr size_t OUT_H = (size_t)T_ * B_ * H_;
  constexpr size_t OUT_C = OUT_H + (size_t)2 * B_ * H_;

  hipFuncSetAttribute(reinterpret_cast<const void*>(k_lstm_rec),
                      hipFuncAttributeMaxDynamicSharedMemorySize, (int)SMEM_REC);

  // 1) conversions
  k_convert_x<<<4096, 256, 0, stream>>>(x, xb);
  k_convert_w<<<4096, 256, 0, stream>>>(Wih0, wi0b, G4_ * H_);
  k_convert_w<<<4096, 256, 0, stream>>>(Whh0, wh0b, G4_ * H_);
  k_convert_w<<<4096, 256, 0, stream>>>(Wih1, wi1b, G4_ * H_);
  k_convert_w<<<4096, 256, 0, stream>>>(Whh1, wh1b, G4_ * H_);
  k_bias_sum<<<G4_ / 256, 256, 0, stream>>>(bih0, bhh0, b0);
  k_bias_sum<<<G4_ / 256, 256, 0, stream>>>(bih1, bhh1, b1);

  // 2) layer 0
  k_gates_gemm<<<dim3(G4_ / 64, T_), 256, 0, stream>>>(xb, wi0b, b0, gxws);
  hipMemsetAsync(ws + HB, 0, (size_t)2 * B_ * H_ * 2, stream);
  hipMemsetAsync(ws + CNT, 0, 256, stream);
  k_lstm_rec<<<NWG_, 256, SMEM_REC, stream>>>(gxws, wh0b, hbuf, y0b, nullptr,
                                              out + OUT_H, out + OUT_C, cnt);

  // 3) layer 1
  k_gates_gemm<<<dim3(G4_ / 64, T_), 256, 0, stream>>>(y0b, wi1b, b1, gxws);
  hipMemsetAsync(ws + HB, 0, (size_t)2 * B_ * H_ * 2, stream);
  hipMemsetAsync(ws + CNT, 0, 256, stream);
  k_lstm_rec<<<NWG_, 256, SMEM_REC, stream>>>(gxws, wh1b, hbuf, nullptr, out,
                                              out + OUT_H + (size_t)B_ * H_,
                                              out + OUT_C + (size_t)B_ * H_, cnt);
}
